// STDANetPP_2860448219491
// MI455X (gfx1250) — compile-verified, matrix operand order checked by reading, not run on hardware
//
#include <hip/hip_runtime.h>
#include <math.h>

typedef __attribute__((ext_vector_type(16))) _Float16 v16h;
typedef __attribute__((ext_vector_type(8)))  _Float16 v8h;
typedef __attribute__((ext_vector_type(8)))  float    v8f;

#define HW   2304
#define WID  48
#define HEI  48

// ---------------------------------------------------------------- helpers ---
struct Taps { int idx[4]; float w[4]; };

__device__ inline Taps mk_taps(float gx, float gy) {
  Taps t;
  float x0f = floorf(gx), y0f = floorf(gy);
  float wx = gx - x0f, wy = gy - y0f;
  int x0 = (int)x0f, y0 = (int)y0f;
  const float ww[4] = {(1.f - wx) * (1.f - wy), wx * (1.f - wy),
                       (1.f - wx) * wy,         wx * wy};
  const int xs[4] = {x0, x0 + 1, x0,     x0 + 1};
  const int ys[4] = {y0, y0,     y0 + 1, y0 + 1};
#pragma unroll
  for (int i = 0; i < 4; ++i) {
    bool v = (xs[i] >= 0) && (xs[i] < WID) && (ys[i] >= 0) && (ys[i] < HEI);
    t.idx[i] = v ? ys[i] * WID + xs[i] : 0;
    t.w[i]   = v ? ww[i] : 0.f;
  }
  return t;
}

__device__ inline float tap_sample(const float* __restrict__ img, const Taps& t) {
  return t.w[0] * img[t.idx[0]] + t.w[1] * img[t.idx[1]] +
         t.w[2] * img[t.idx[2]] + t.w[3] * img[t.idx[3]];
}

// ---------------------------------------------------- weight pack f32->f16 ---
__global__ void k_packw(const float* __restrict__ s, _Float16* __restrict__ d,
                        int Cout, int K, int Kp) {
  int i = blockIdx.x * 256 + threadIdx.x;
  if (i >= Cout * Kp) return;
  int r = i / Kp, k = i - r * Kp;
  d[i] = (k < K) ? (_Float16)s[r * K + k] : (_Float16)0.f;
}

// ----------------------------------------------------------- flow chaining ---
__global__ void k_flow(const float* __restrict__ f1, const float* __restrict__ f2,
                       float* __restrict__ fcn) {
  int q = blockIdx.x * 256 + threadIdx.x;
  if (q >= HW) return;
  float qx = (float)(q % WID), qy = (float)(q / WID);
  float ax = f1[q], ay = f1[HW + q];
  Taps t = mk_taps(qx + ax, qy + ay);
  float bx = ax + tap_sample(f2, t);
  float by = ay + tap_sample(f2 + HW, t);
  fcn[q] = ax; fcn[HW + q] = ay; fcn[2 * HW + q] = bx; fcn[3 * HW + q] = by;
}

// -------------------- build extra[HW][224] and xcat[HW][192] (pixel-major) ---
__global__ void k_build(const float* __restrict__ x, const float* __restrict__ fcn,
                        _Float16* __restrict__ extra, _Float16* __restrict__ xcat) {
  int q = blockIdx.x * 256 + threadIdx.x;
  if (q >= HW) return;
  _Float16* xr = xcat + (size_t)q * 192;
  _Float16* er = extra + (size_t)q * 224;
  for (int c = 0; c < 192; ++c) xr[c] = (_Float16)x[c * HW + q];
  for (int c = 0; c < 64;  ++c) er[c] = (_Float16)x[c * HW + q];
  float qx = (float)(q % WID), qy = (float)(q / WID);
  float f1x = fcn[q], f1y = fcn[HW + q], f2x = fcn[2 * HW + q], f2y = fcn[3 * HW + q];
  Taps t1 = mk_taps(qx + f1x, qy + f1y);
  Taps t2 = mk_taps(qx + f2x, qy + f2y);
  const float* x1 = x + 64 * HW;
  const float* x2 = x + 128 * HW;
  for (int c = 0; c < 64; ++c) er[64 + c]  = (_Float16)tap_sample(x1 + c * HW, t1);
  for (int c = 0; c < 64; ++c) er[128 + c] = (_Float16)tap_sample(x2 + c * HW, t2);
  er[192] = (_Float16)f1x; er[193] = (_Float16)f1y;
  er[194] = (_Float16)f2x; er[195] = (_Float16)f2y;
  for (int c = 196; c < 224; ++c) er[c] = (_Float16)0.f;
}

// -------------------------- im2col 3x3 pad=1 : in[HW][64] -> out[HW][576] ----
__global__ void k_im2col(const _Float16* __restrict__ in, _Float16* __restrict__ out) {
  int i = blockIdx.x * 256 + threadIdx.x;
  if (i >= HW * 576) return;
  int q = i / 576, k = i - q * 576;
  int c = k / 9, t = k - c * 9;
  int y = q / WID + t / 3 - 1;
  int xx = q % WID + t % 3 - 1;
  _Float16 v = (_Float16)0.f;
  if (y >= 0 && y < HEI && xx >= 0 && xx < WID) v = in[(size_t)(y * WID + xx) * 64 + c];
  out[i] = v;
}

// ------------------------------------------------------ WMMA GEMM (wave32) ---
// D[q][m] = sum_k A[m][k] * Bt[q][k] + bias[m].  A row-major [M][Kp],
// activations pixel-major [HW][Kp] -> both fragments are two contiguous
// 16B global_load_b128 per lane.  One wave = 32x32 output super-tile
// (2 M-tiles x 2 N-tiles, 4 WMMAs per K-step).  Output pixel-major [HW][M].
template <int ACT, int OUTF16>
__global__ void gemm_wmma(const _Float16* __restrict__ A, const _Float16* __restrict__ Bt,
                          const float* __restrict__ bias, void* __restrict__ out,
                          int M, int Kp) {
  const int lane = threadIdx.x;
  const int nl   = lane & 15;
  const int grp  = lane >> 4;
  const int koff = grp * 8;
  const int n0   = blockIdx.x * 32;
  const int m0   = blockIdx.y * 32;
  const _Float16* a0p = A  + (size_t)(m0 + nl)      * Kp + koff;
  const _Float16* a1p = A  + (size_t)(m0 + 16 + nl) * Kp + koff;
  const _Float16* b0p = Bt + (size_t)(n0 + nl)      * Kp + koff;
  const _Float16* b1p = Bt + (size_t)(n0 + 16 + nl) * Kp + koff;
  v8f acc00 = {}, acc01 = {}, acc10 = {}, acc11 = {};
  for (int k0 = 0; k0 < Kp; k0 += 32) {
    v16h a0, a1, b0, b1;
    ((v8h*)&a0)[0] = *(const v8h*)(a0p + k0);
    ((v8h*)&a0)[1] = *(const v8h*)(a0p + k0 + 16);
    ((v8h*)&a1)[0] = *(const v8h*)(a1p + k0);
    ((v8h*)&a1)[1] = *(const v8h*)(a1p + k0 + 16);
    ((v8h*)&b0)[0] = *(const v8h*)(b0p + k0);
    ((v8h*)&b0)[1] = *(const v8h*)(b0p + k0 + 16);
    ((v8h*)&b1)[0] = *(const v8h*)(b1p + k0);
    ((v8h*)&b1)[1] = *(const v8h*)(b1p + k0 + 16);
    acc00 = __builtin_amdgcn_wmma_f32_16x16x32_f16(false, a0, false, b0, (short)0, acc00, false, false);
    acc01 = __builtin_amdgcn_wmma_f32_16x16x32_f16(false, a0, false, b1, (short)0, acc01, false, false);
    acc10 = __builtin_amdgcn_wmma_f32_16x16x32_f16(false, a1, false, b0, (short)0, acc10, false, false);
    acc11 = __builtin_amdgcn_wmma_f32_16x16x32_f16(false, a1, false, b1, (short)0, acc11, false, false);
  }
  auto emit = [&](const v8f& acc, int mi, int ni) {
    int q  = n0 + ni * 16 + nl;
    int mb = m0 + mi * 16 + grp * 8;
    if (OUTF16) {
      v8h pv;
#pragma unroll
      for (int r = 0; r < 8; ++r) {
        float v = acc[r] + bias[mb + r];
        if (ACT) v = (v >= 0.f) ? v : 0.1f * v;
        pv[r] = (_Float16)v;
      }
      *(v8h*)((_Float16*)out + (size_t)q * M + mb) = pv;
    } else {
      v8f pv;
#pragma unroll
      for (int r = 0; r < 8; ++r) {
        float v = acc[r] + bias[mb + r];
        if (ACT) v = (v >= 0.f) ? v : 0.1f * v;
        pv[r] = v;
      }
      *(v8f*)((float*)out + (size_t)q * M + mb) = pv;
    }
  };
  emit(acc00, 0, 0); emit(acc01, 0, 1); emit(acc10, 1, 0); emit(acc11, 1, 1);
}

// ------------------------------------------- deformable attention sampling ---
// thread t: q = t % HW, h = t / HW.  aw[HW][2592], so[HW][576], po[HW][288],
// value[HW][192] (channel = (l*8+h)*8+d), fcn[4][HW].  out f16 [HW][64].
__global__ void k_attn(const float* __restrict__ aw, const float* __restrict__ so,
                       const float* __restrict__ po, const float* __restrict__ fcn,
                       const float* __restrict__ value, _Float16* __restrict__ outh) {
  int t = blockIdx.x * 256 + threadIdx.x;
  if (t >= 8 * HW) return;
  int q = t % HW, h = t / HW;
  float qx = (float)(q % WID), qy = (float)(q / WID);
  float flx[3] = {0.f, fcn[q],      fcn[2 * HW + q]};
  float fly[3] = {0.f, fcn[HW + q], fcn[3 * HW + q]};
  const float* awq = aw + (size_t)q * 2592 + (size_t)h * 324;
  const float* soq = so + (size_t)q * 576;
  const float* poq = po + (size_t)q * 288;
  float mx = -1e30f;
  for (int j = 0; j < 324; ++j) mx = fmaxf(mx, awq[j]);
  float se = 0.f;
  for (int j = 0; j < 324; ++j) se += expf(awq[j] - mx);
  float inv = 1.f / se;
  float acc[8] = {0.f, 0.f, 0.f, 0.f, 0.f, 0.f, 0.f, 0.f};
  for (int l = 0; l < 3; ++l) {
    int vbase = (l * 8 + h) * 8;
    for (int p = 0; p < 12; ++p) {
      int ch = (h * 3 + l) * 12 + p;
      float sx = soq[ch * 2 + 0];
      float sy = soq[ch * 2 + 1];
      float ps = 3.f / (1.f + expf(-poq[ch]));
      for (int cc = 0; cc < 9; ++cc) {
        float gx = qx + sx + flx[l] + (float)(cc / 3 - 1) * ps;
        float gy = qy + sy + fly[l] + (float)(cc % 3 - 1) * ps;
        int j = (l * 12 + p) * 9 + cc;
        float w = expf(awq[j] - mx) * inv;
        Taps tp = mk_taps(gx, gy);
#pragma unroll
        for (int i = 0; i < 4; ++i) {
          float ww = w * tp.w[i];
          if (ww != 0.f) {
            const float* vrow = value + (size_t)tp.idx[i] * 192 + vbase;
#pragma unroll
            for (int d = 0; d < 8; ++d) acc[d] += ww * vrow[d];
          }
        }
      }
    }
  }
  _Float16* orow = outh + (size_t)q * 64 + h * 8;
  for (int d = 0; d < 8; ++d) orow[d] = (_Float16)acc[d];
}

// -------------------------- residual + transpose [HW][64] -> NCHW [64][HW] ---
__global__ void k_resid(const float* __restrict__ y, const float* __restrict__ x0,
                        float* __restrict__ out, int n) {
  int i = blockIdx.x * 256 + threadIdx.x;
  if (i >= n) return;
  int c = i / HW, q = i - c * HW;
  out[i] = y[(size_t)q * 64 + c] + x0[i];
}

// ----------------------------------------------------------------- launch ---
extern "C" void kernel_launch(void* const* d_in, const int* in_sizes, int n_in,
                              void* d_out, int out_size, void* d_ws, size_t ws_size,
                              hipStream_t stream) {
  (void)in_sizes; (void)n_in; (void)out_size; (void)ws_size;
  const float* x  = (const float*)d_in[0];
  const float* f1 = (const float*)d_in[1];
  const float* f2 = (const float*)d_in[2];
  const float* bw[3][8];  // dict order: aw, so, po ; w1,b1,w2,b2,w3,b3,w4,b4
  int idx = 3;
  for (int b = 0; b < 3; ++b)
    for (int j = 0; j < 8; ++j) bw[b][j] = (const float*)d_in[idx++];
  const float* v_w  = (const float*)d_in[idx++];
  const float* v_b  = (const float*)d_in[idx++];
  const float* o_w1 = (const float*)d_in[idx++];
  const float* o_b1 = (const float*)d_in[idx++];
  const float* o_w2 = (const float*)d_in[idx++];
  const float* o_b2 = (const float*)d_in[idx++];
  const int co[3] = {2592, 576, 288};

  char* ws = (char*)d_ws;
  size_t off = 0;
  auto alloc = [&](size_t bytes) -> char* {
    char* p = ws + off;
    off += (bytes + 255) & ~(size_t)255;
    return p;
  };
  _Float16* xcat  = (_Float16*)alloc((size_t)HW * 192 * 2);
  _Float16* extra = (_Float16*)alloc((size_t)HW * 224 * 2);
  float*    fcn   = (float*)alloc((size_t)4 * HW * 4);
  float*    value = (float*)alloc((size_t)HW * 192 * 4);
  _Float16* yA    = (_Float16*)alloc((size_t)HW * 64 * 2);
  _Float16* yB    = (_Float16*)alloc((size_t)HW * 64 * 2);
  _Float16* col   = (_Float16*)alloc((size_t)HW * 576 * 2);
  float* bro[3];
  bro[0] = (float*)alloc((size_t)HW * 2592 * 4);
  bro[1] = (float*)alloc((size_t)HW * 576 * 4);
  bro[2] = (float*)alloc((size_t)HW * 288 * 4);
  _Float16* attnh = (_Float16*)alloc((size_t)HW * 64 * 2);
  _Float16* midh  = (_Float16*)alloc((size_t)HW * 64 * 2);
  float*    out2  = (float*)alloc((size_t)HW * 64 * 4);
  _Float16* vwp   = (_Float16*)alloc((size_t)192 * 192 * 2);
  _Float16 *w1p[3], *w2p[3], *w3p[3], *w4p[3];
  for (int b = 0; b < 3; ++b) {
    w1p[b] = (_Float16*)alloc((size_t)64 * 224 * 2);
    w2p[b] = (_Float16*)alloc((size_t)64 * 576 * 2);
    w3p[b] = (_Float16*)alloc((size_t)64 * 576 * 2);
    w4p[b] = (_Float16*)alloc((size_t)co[b] * 64 * 2);
  }
  _Float16* ow1p = (_Float16*)alloc((size_t)64 * 64 * 2);
  _Float16* ow2p = (_Float16*)alloc((size_t)64 * 64 * 2);

  auto pk = [&](const float* s, _Float16* d, int Cout, int K, int Kp) {
    int n = Cout * Kp;
    k_packw<<<(n + 255) / 256, 256, 0, stream>>>(s, d, Cout, K, Kp);
  };
  pk(v_w, vwp, 192, 192, 192);
  for (int b = 0; b < 3; ++b) {
    pk(bw[b][0], w1p[b], 64, 196, 224);
    pk(bw[b][2], w2p[b], 64, 576, 576);
    pk(bw[b][4], w3p[b], 64, 576, 576);
    pk(bw[b][6], w4p[b], co[b], 64, 64);
  }
  pk(o_w1, ow1p, 64, 64, 64);
  pk(o_w2, ow2p, 64, 64, 64);

  k_flow<<<(HW + 255) / 256, 256, 0, stream>>>(f1, f2, fcn);
  k_build<<<(HW + 255) / 256, 256, 0, stream>>>(x, fcn, extra, xcat);

  dim3 wb(32);
  const int NT = HW / 32;  // 72 pixel super-tiles
  gemm_wmma<0, 0><<<dim3(NT, 192 / 32), wb, 0, stream>>>(vwp, xcat, v_b, value, 192, 192);
  for (int b = 0; b < 3; ++b) {
    gemm_wmma<1, 1><<<dim3(NT, 2), wb, 0, stream>>>(w1p[b], extra, bw[b][1], yA, 64, 224);
    k_im2col<<<(HW * 576 + 255) / 256, 256, 0, stream>>>(yA, col);
    gemm_wmma<1, 1><<<dim3(NT, 2), wb, 0, stream>>>(w2p[b], col, bw[b][3], yB, 64, 576);
    k_im2col<<<(HW * 576 + 255) / 256, 256, 0, stream>>>(yB, col);
    gemm_wmma<1, 1><<<dim3(NT, 2), wb, 0, stream>>>(w3p[b], col, bw[b][5], yA, 64, 576);
    gemm_wmma<0, 0><<<dim3(NT, co[b] / 32), wb, 0, stream>>>(w4p[b], yA, bw[b][7], bro[b],
                                                             co[b], 64);
  }
  k_attn<<<(8 * HW + 255) / 256, 256, 0, stream>>>(bro[0], bro[1], bro[2], fcn, value, attnh);
  gemm_wmma<1, 1><<<dim3(NT, 2), wb, 0, stream>>>(ow1p, attnh, o_b1, midh, 64, 64);
  gemm_wmma<0, 0><<<dim3(NT, 2), wb, 0, stream>>>(ow2p, midh, o_b2, out2, 64, 64);
  k_resid<<<(64 * HW + 255) / 256, 256, 0, stream>>>(out2, x, (float*)d_out, 64 * HW);
}
